// GATEncoder_6640019439959
// MI455X (gfx1250) — matrix operand order repaired and checked by reading, not compiled, not run on hardware
//
#include <hip/hip_runtime.h>
#include <hip/hip_bf16.h>

#define NN 50000
#define EE 800000
#define EP_ 850000            // EE + NN self loops
#define GG 128
#define OUT_STRIDE 576        // 256 + 256 + 64
#define LDSK 264              // 256 max K + 8 bf16 pad (bank-conflict-free)

typedef __attribute__((ext_vector_type(16))) __bf16 v16bf;
typedef __attribute__((ext_vector_type(8)))  __bf16 v8bf;
typedef __attribute__((ext_vector_type(8)))  float  v8f;

// ---------- monotonic float <-> uint for atomicMax over floats ----------
static __device__ __forceinline__ unsigned f2ord(float f) {
  unsigned b = __float_as_uint(f);
  return (b & 0x80000000u) ? ~b : (b | 0x80000000u);
}
static __device__ __forceinline__ float ord2f(unsigned u) {
  unsigned b = (u & 0x80000000u) ? (u ^ 0x80000000u) : ~u;
  return __uint_as_float(b);
}
#define ORD_NEG_INF 0x007fffffu   // f2ord(-inf)

// ---------- small utility kernels ----------
__global__ void k_zero_f32(float* __restrict__ p, long long n) {
  long long i = (long long)blockIdx.x * blockDim.x + threadIdx.x;
  if (i < n) p[i] = 0.0f;
}
__global__ void k_fill_u32(unsigned* __restrict__ p, unsigned v, long long n) {
  long long i = (long long)blockIdx.x * blockDim.x + threadIdx.x;
  if (i < n) p[i] = v;
}
__global__ void k_f2bf(const float* __restrict__ x, __bf16* __restrict__ o, long long n) {
  long long i = (long long)blockIdx.x * blockDim.x + threadIdx.x;
  if (i < n) o[i] = (__bf16)x[i];
}
// W [K, Nout] fp32  ->  Wt [Nout, K] bf16  (gives contiguous-K B fragments)
__global__ void k_wt(const float* __restrict__ W, __bf16* __restrict__ Wt, int K, int Nout) {
  int idx = blockIdx.x * blockDim.x + threadIdx.x;
  if (idx >= K * Nout) return;
  int k = idx / Nout, n = idx % Nout;
  Wt[(size_t)n * K + k] = (__bf16)W[idx];
}

// ---------- bf16 WMMA GEMM:  C[M,Nout] = A[M,K] * Wt[Nout,K]^T ----------
// Block: 256 threads = 8 waves, covers 128 rows x (NT*16) cols.
// Wt tile staged in LDS once per block; each wave: 16 rows x NT*16 cols,
// NT WMMAs per A fetch, B fragments via ds_load_b128.
template<int NT>
__global__ __launch_bounds__(256) void k_gemm_bf16(
    const __bf16* __restrict__ A, const __bf16* __restrict__ Bt,
    float* __restrict__ C, int M, int K, int Nout) {
  __shared__ __bf16 lb[NT * 16][LDSK];

  const int lane = threadIdx.x & 31;
  const int wave = threadIdx.x >> 5;
  const int n0   = blockIdx.y * (NT * 16);

  // cooperative staging of Wt[n0 .. n0+NT*16) rows into LDS (8-elem chunks)
  {
    const int kc = K >> 3;                    // chunks per row
    const int nchunks = NT * 16 * kc;
    for (int idx = threadIdx.x; idx < nchunks; idx += 256) {
      int col = idx / kc;
      int ko  = (idx - col * kc) << 3;
      *(v8bf*)&lb[col][ko] = *(const v8bf*)(Bt + (size_t)(n0 + col) * K + ko);
    }
  }
  __syncthreads();

  const int m0 = (blockIdx.x * 8 + wave) << 4;
  if (m0 >= M) return;                        // wave-uniform exit (post-barrier)

  const __bf16* ap = A + (size_t)(m0 + (lane & 15)) * K + ((lane >> 4) << 4);
  const int bcol = lane & 15;                 // local column within LDS tile
  const int bko  = (lane >> 4) << 4;

  v8f zero;
#pragma unroll
  for (int i = 0; i < 8; ++i) zero[i] = 0.0f;
  v8f acc[NT];
#pragma unroll
  for (int t = 0; t < NT; ++t) acc[t] = zero;

  for (int k0 = 0; k0 < K; k0 += 32) {
    union { v16bf v; v8bf h[2]; } ua;
    ua.h[0] = *(const v8bf*)(ap + k0);        // K = kb .. kb+7
    ua.h[1] = *(const v8bf*)(ap + k0 + 8);    // K = kb+8 .. kb+15
    __builtin_prefetch((const void*)(ap + k0 + 32), 0, 1);
#pragma unroll
    for (int t = 0; t < NT; ++t) {
      const __bf16* bp = &lb[t * 16 + bcol][k0 + bko];
      union { v16bf v; v8bf h[2]; } ub;
      ub.h[0] = *(const v8bf*)(bp);
      ub.h[1] = *(const v8bf*)(bp + 8);
      acc[t] = __builtin_amdgcn_wmma_f32_16x16x32_bf16(
          false, ua.v, false, ub.v, (short)0, acc[t], false, false);
    }
  }
  const int crow0 = m0 + ((lane >> 4) << 3);
  const int ccol  = n0 + (lane & 15);
#pragma unroll
  for (int t = 0; t < NT; ++t)
#pragma unroll
    for (int v = 0; v < 8; ++v)
      C[(size_t)(crow0 + v) * Nout + (ccol + (t << 4))] = acc[t][v];
}

// ---------- per-(node,head) attention dot products ----------
__global__ void k_attn(const float* __restrict__ hb, const float* __restrict__ asrc,
                       const float* __restrict__ adst, float* __restrict__ as_,
                       float* __restrict__ ad_, int heads, int C) {
  int idx = blockIdx.x * blockDim.x + threadIdx.x;   // idx = n*heads + h
  if (idx >= NN * heads) return;
  int h = idx % heads;
  const float* hp = hb + (size_t)idx * C;            // [N, heads, C] layout
  const float* pa = asrc + h * C;
  const float* pd = adst + h * C;
  float sa = 0.f, sd = 0.f;
  for (int c = 0; c < C; c += 4) {
    float4 hv = *(const float4*)(hp + c);
    float4 av = *(const float4*)(pa + c);
    float4 dv = *(const float4*)(pd + c);
    sa += hv.x * av.x + hv.y * av.y + hv.z * av.z + hv.w * av.w;
    sd += hv.x * dv.x + hv.y * dv.y + hv.z * dv.z + hv.w * dv.w;
  }
  as_[idx] = sa;
  ad_[idx] = sd;
}

// ---------- edge pass A: segment max of leaky-relu score ----------
__global__ void k_edge_max(const int* __restrict__ ei, const float* __restrict__ as_,
                           const float* __restrict__ ad_, unsigned* __restrict__ mbuf,
                           int heads) {
  int e = blockIdx.x * blockDim.x + threadIdx.x;
  if (e >= EP_) return;
  int s, d;
  if (e < EE) { s = ei[e]; d = ei[EE + e]; } else { s = d = e - EE; }
  for (int h = 0; h < heads; ++h) {
    float v = as_[s * heads + h] + ad_[d * heads + h];
    v = v > 0.f ? v : 0.2f * v;
    atomicMax(&mbuf[d * heads + h], f2ord(v));
  }
}

// ---------- edge pass B: exp(score - max), segment sum ----------
__global__ void k_edge_expsum(const int* __restrict__ ei, const float* __restrict__ as_,
                              const float* __restrict__ ad_, const unsigned* __restrict__ mbuf,
                              float* __restrict__ eexp, float* __restrict__ ssum, int heads) {
  int e = blockIdx.x * blockDim.x + threadIdx.x;
  if (e >= EP_) return;
  int s, d;
  if (e < EE) { s = ei[e]; d = ei[EE + e]; } else { s = d = e - EE; }
  for (int h = 0; h < heads; ++h) {
    float v = as_[s * heads + h] + ad_[d * heads + h];
    v = v > 0.f ? v : 0.2f * v;
    float m = ord2f(mbuf[d * heads + h]);
    float ex = __expf(v - m);
    eexp[(size_t)e * heads + h] = ex;
    atomicAdd(&ssum[d * heads + h], ex);
  }
}

// ---------- edge pass C: weighted scatter-add, one wave per edge ----------
__global__ __launch_bounds__(256) void k_edge_agg(
    const int* __restrict__ ei, const float* __restrict__ hbuf,
    const float* __restrict__ eexp, const float* __restrict__ ssum,
    float* __restrict__ agg, int heads, int C) {
  int w    = (int)((blockIdx.x * (size_t)blockDim.x + threadIdx.x) >> 5);
  int lane = threadIdx.x & 31;
  if (w >= EP_) return;
  int s, d;
  if (w < EE) { s = ei[w]; d = ei[EE + w]; } else { s = d = w - EE; }
  int HC  = heads * C;
  int cpl = HC >> 5;                 // channels per lane: 8 (H=4) or 2 (H=1)
  int ch0 = lane * cpl;
  int hh  = ch0 / C;
  float alpha = eexp[(size_t)w * heads + hh] / (ssum[d * heads + hh] + 1e-16f);
  const float* hp = hbuf + (size_t)s * HC + ch0;
  float* op = agg + (size_t)d * HC + ch0;
  if (cpl == 8) {
    float4 a0 = *(const float4*)(hp);
    float4 a1 = *(const float4*)(hp + 4);
    atomicAdd(op + 0, alpha * a0.x); atomicAdd(op + 1, alpha * a0.y);
    atomicAdd(op + 2, alpha * a0.z); atomicAdd(op + 3, alpha * a0.w);
    atomicAdd(op + 4, alpha * a1.x); atomicAdd(op + 5, alpha * a1.y);
    atomicAdd(op + 6, alpha * a1.z); atomicAdd(op + 7, alpha * a1.w);
  } else {
    float2 a0 = *(const float2*)(hp);
    atomicAdd(op + 0, alpha * a0.x); atomicAdd(op + 1, alpha * a0.y);
  }
}

// ---------- bias + ELU ----------
__global__ void k_bias_elu(const float* __restrict__ agg, const float* __restrict__ bias,
                           float* __restrict__ act, int C) {
  long long i = (long long)blockIdx.x * blockDim.x + threadIdx.x;
  if (i >= (long long)NN * C) return;
  int c = (int)(i % C);
  float v = agg[i] + bias[c];
  act[i] = v > 0.f ? v : (__expf(v) - 1.f);
}

// ---------- batchnorm: column sums (coalesced across channels) ----------
__global__ void k_bn_stats(const float* __restrict__ act, float* __restrict__ stats, int C) {
  int c  = threadIdx.x;              // blockDim.x == C
  int r0 = blockIdx.x * 128;
  int r1 = r0 + 128; if (r1 > NN) r1 = NN;
  float s = 0.f, s2 = 0.f;
  for (int r = r0; r < r1; ++r) {
    float v = act[(size_t)r * C + c];
    s += v; s2 += v * v;
  }
  atomicAdd(&stats[c], s);
  atomicAdd(&stats[C + c], s2);
}
__global__ void k_bn_final(const float* __restrict__ stats, const float* __restrict__ gamma,
                           const float* __restrict__ beta, float* __restrict__ sc,
                           float* __restrict__ sh, int C) {
  int c = threadIdx.x;
  if (c >= C) return;
  float mu   = stats[c] / (float)NN;
  float var  = stats[C + c] / (float)NN - mu * mu;
  float g    = gamma[c] * rsqrtf(var + 1e-5f);
  sc[c] = g;
  sh[c] = beta[c] - mu * g;
}
// normalize; write fp32 xs slice of d_out and bf16 input for next layer
__global__ void k_bn_apply(const float* __restrict__ act, const float* __restrict__ sc,
                           const float* __restrict__ sh, float* __restrict__ xs,
                           __bf16* __restrict__ xb_next, int C, int loff, int write_bf) {
  long long i = (long long)blockIdx.x * blockDim.x + threadIdx.x;
  if (i >= (long long)NN * C) return;
  int c = (int)(i % C);
  long long n = i / C;
  float y = act[i] * sc[c] + sh[c];
  xs[n * OUT_STRIDE + loff + c] = y;
  if (write_bf) xb_next[i] = (__bf16)y;
}

// ---------- graph pooling over sorted batch ids ----------
__global__ void k_pool(const float* __restrict__ xs, const int* __restrict__ batch,
                       float* __restrict__ pool) {
  int c  = threadIdx.x;              // 576 threads
  int r0 = blockIdx.x * 256;
  if (r0 >= NN) return;
  int r1 = r0 + 256; if (r1 > NN) r1 = NN;
  float acc = 0.f;
  int g = batch[r0];
  for (int r = r0; r < r1; ++r) {
    int gr = batch[r];
    if (gr != g) { atomicAdd(&pool[(size_t)g * OUT_STRIDE + c], acc); acc = 0.f; g = gr; }
    acc += xs[(size_t)r * OUT_STRIDE + c];
  }
  atomicAdd(&pool[(size_t)g * OUT_STRIDE + c], acc);
}

// =====================================================================
extern "C" void kernel_launch(void* const* d_in, const int* in_sizes, int n_in,
                              void* d_out, int out_size, void* d_ws, size_t ws_size,
                              hipStream_t stream) {
  (void)in_sizes; (void)n_in; (void)out_size; (void)ws_size;
  const float* x     = (const float*)d_in[0];
  const int*   ei    = (const int*)d_in[1];
  const int*   batch = (const int*)d_in[2];

  struct L { const float *W, *asrc, *adst, *b, *gamma, *beta; int K, heads, C, loff; };
  L ls[3] = {
    { (const float*)d_in[3],  (const float*)d_in[4],  (const float*)d_in[5],
      (const float*)d_in[6],  (const float*)d_in[7],  (const float*)d_in[8],  128, 4, 64, 0 },
    { (const float*)d_in[9],  (const float*)d_in[10], (const float*)d_in[11],
      (const float*)d_in[12], (const float*)d_in[13], (const float*)d_in[14], 256, 4, 64, 256 },
    { (const float*)d_in[15], (const float*)d_in[16], (const float*)d_in[17],
      (const float*)d_in[18], (const float*)d_in[19], (const float*)d_in[20], 256, 1, 64, 512 },
  };

  // workspace partition (256B aligned)
  char* base = (char*)d_ws;
  size_t off = 0;
  auto take = [&](size_t bytes) -> void* {
    void* p = base + off;
    off += (bytes + 255) & ~(size_t)255;
    return p;
  };
  __bf16*  xb    = (__bf16*)take((size_t)NN * 256 * 2);
  __bf16*  wt0   = (__bf16*)take((size_t)128 * 256 * 2);
  __bf16*  wt1   = (__bf16*)take((size_t)256 * 256 * 2);
  __bf16*  wt2   = (__bf16*)take((size_t)256 * 64 * 2);
  float*   hbuf  = (float*)take((size_t)NN * 256 * 4);   // GEMM out, reused as 'act'
  float*   agg   = (float*)take((size_t)NN * 256 * 4);
  float*   as_   = (float*)take((size_t)NN * 4 * 4);
  float*   ad_   = (float*)take((size_t)NN * 4 * 4);
  unsigned* mbuf = (unsigned*)take((size_t)NN * 4 * 4);
  float*   ssum  = (float*)take((size_t)NN * 4 * 4);
  float*   eexp  = (float*)take((size_t)EP_ * 4 * 4);
  float*   stats = (float*)take((size_t)512 * 4);
  float*   scale = (float*)take((size_t)256 * 4);
  float*   shift = (float*)take((size_t)256 * 4);
  __bf16*  wts[3] = { wt0, wt1, wt2 };

  float* pool    = (float*)d_out;
  float* xs_base = (float*)d_out + (size_t)GG * OUT_STRIDE;

  // weight prep + layer-0 input conversion
  k_wt  <<<(128 * 256 + 255) / 256, 256, 0, stream>>>(ls[0].W, wt0, 128, 256);
  k_wt  <<<(256 * 256 + 255) / 256, 256, 0, stream>>>(ls[1].W, wt1, 256, 256);
  k_wt  <<<(256 * 64  + 255) / 256, 256, 0, stream>>>(ls[2].W, wt2, 256, 64);
  k_f2bf<<<((long long)NN * 128 + 255) / 256, 256, 0, stream>>>(x, xb, (long long)NN * 128);

  for (int li = 0; li < 3; ++li) {
    const L& L_ = ls[li];
    const int heads = L_.heads, C = L_.C, K = L_.K;
    const int Nout  = heads * C;                 // 256, 256, 64
    const long long nNout = (long long)NN * Nout;

    const int mblocks = (NN / 16 + 7) / 8;       // 391
    if (Nout == 256) {
      dim3 gg(mblocks, 2);                       // 128-col LDS tiles
      k_gemm_bf16<8><<<gg, 256, 0, stream>>>(xb, wts[li], hbuf, NN, K, Nout);
    } else {
      dim3 gg(mblocks, 1);                       // 64-col LDS tile
      k_gemm_bf16<4><<<gg, 256, 0, stream>>>(xb, wts[li], hbuf, NN, K, Nout);
    }

    k_attn<<<(NN * heads + 255) / 256, 256, 0, stream>>>(
        hbuf, L_.asrc, L_.adst, as_, ad_, heads, C);

    k_fill_u32<<<(NN * heads + 255) / 256, 256, 0, stream>>>(
        mbuf, ORD_NEG_INF, (long long)NN * heads);
    k_zero_f32<<<(NN * heads + 255) / 256, 256, 0, stream>>>(ssum, (long long)NN * heads);
    k_zero_f32<<<(nNout + 255) / 256, 256, 0, stream>>>(agg, nNout);

    k_edge_max   <<<(EP_ + 255) / 256, 256, 0, stream>>>(ei, as_, ad_, mbuf, heads);
    k_edge_expsum<<<(EP_ + 255) / 256, 256, 0, stream>>>(ei, as_, ad_, mbuf, eexp, ssum, heads);
    k_edge_agg   <<<(EP_ + 7) / 8,     256, 0, stream>>>(ei, hbuf, eexp, ssum, agg, heads, C);

    // bias + ELU (act overwrites hbuf, which is dead after edge_agg)
    k_bias_elu<<<(nNout + 255) / 256, 256, 0, stream>>>(agg, L_.b, hbuf, Nout);

    // batchnorm (training stats)
    k_zero_f32<<<(2 * Nout + 255) / 256, 256, 0, stream>>>(stats, 2 * Nout);
    k_bn_stats<<<(NN + 127) / 128, Nout, 0, stream>>>(hbuf, stats, Nout);
    k_bn_final<<<1, Nout, 0, stream>>>(stats, L_.gamma, L_.beta, scale, shift, Nout);
    k_bn_apply<<<(nNout + 255) / 256, 256, 0, stream>>>(
        hbuf, scale, shift, xs_base, xb, Nout, L_.loff, li < 2 ? 1 : 0);
  }

  // pooled output
  k_zero_f32<<<((long long)GG * OUT_STRIDE + 255) / 256, 256, 0, stream>>>(
      pool, (long long)GG * OUT_STRIDE);
  k_pool<<<(NN + 255) / 256, OUT_STRIDE, 0, stream>>>(xs_base, batch, pool);
}
